// VoxelNeRFBase_80736795230827
// MI455X (gfx1250) — matrix operand order repaired and checked by reading, not compiled, not run on hardware
//
#include <hip/hip_runtime.h>
#include <hip/hip_bf16.h>

typedef __attribute__((ext_vector_type(16))) _Float16 v16h;
typedef __attribute__((ext_vector_type(8)))  float    v8f;

#define GRID_SZ   512
#define C_TOTAL   96      // 64 + 16 + 16
#define APP_DIM   32
#define WAVES_PB  8       // 8 wave32 per 256-thread block
#define FEAT_STRIDE 98    // halfs per row; 49 dwords, gcd(49,64)=1 -> conflict-free
#define AABB_MIN  (-1.5f)
#define AABB_INV  (1.0f/3.0f)

// 16-bit A-matrix 16x32 VGPR layout (ISA 7.12.2):
// lanes 0-15:  j=0..7 -> K=j,      j=8..15 -> K=16+(j-8)
// lanes 16-31: j=0..7 -> K=8+j,    j=8..15 -> K=24+(j-8)
__device__ __forceinline__ int ka_idx(int lane, int j) {
    int hi = (lane >> 4) & 1;
    return (j < 8) ? ((hi ? 8 : 0) + j) : ((hi ? 24 : 16) + (j - 8));
}
// contiguous k order used here for both A and B: lanes 0-15 K=0..15, lanes 16-31 K=16..31, N=lane%16
__device__ __forceinline__ int kb_idx(int lane, int j) {
    return (((lane >> 4) & 1) ? 16 : 0) + j;
}

__global__ __launch_bounds__(256, 2)
void tensorf_app_kernel(const float* __restrict__ pts,
                        const float* __restrict__ plane0,
                        const float* __restrict__ plane1,
                        const float* __restrict__ plane2,
                        const float* __restrict__ line0,
                        const float* __restrict__ line1,
                        const float* __restrict__ line2,
                        const float* __restrict__ basis_W,  // (32, 96)
                        float* __restrict__ out,            // (N, 32)
                        int n_points) {
    __shared__ _Float16 s_feat[WAVES_PB][16][FEAT_STRIDE];
    __shared__ int      s_i0[WAVES_PB][16][3];
    __shared__ int      s_i1[WAVES_PB][16][3];
    __shared__ float    s_w [WAVES_PB][16][3];

    const int tid    = threadIdx.x;
    const int lane   = tid & 31;
    const int wv     = tid >> 5;
    const int n_col  = lane & 15;

    // ---- Load B (basis_W transposed view: B[k][n] = W[n][k]) once, f32->f16 ----
    v16h bmat[3][2];
    #pragma unroll
    for (int kc = 0; kc < 3; ++kc)
        #pragma unroll
        for (int nt = 0; nt < 2; ++nt) {
            v16h t;
            #pragma unroll
            for (int j = 0; j < 16; ++j) {
                int k = kc * 32 + kb_idx(lane, j);
                t[j] = (_Float16)basis_W[(size_t)(nt * 16 + n_col) * C_TOTAL + k];
            }
            bmat[kc][nt] = t;
        }

    // ---- Per-lane static gather config for each 32-channel chunk ----
    // chunk 0,1: plane0 (64 ch). chunk 2: lanes 0-15 plane1, lanes 16-31 plane2.
    const float* g_plane[3]; const float* g_line[3];
    int g_C[3], g_c[3], g_a0[3], g_a1[3], g_av[3];
    #pragma unroll
    for (int cc = 0; cc < 3; ++cc) {
        int ch = cc * 32 + lane;
        if (ch < 64)      { g_plane[cc]=plane0; g_line[cc]=line0; g_C[cc]=64; g_c[cc]=ch;    g_a0[cc]=0; g_a1[cc]=1; g_av[cc]=2; }
        else if (ch < 80) { g_plane[cc]=plane1; g_line[cc]=line1; g_C[cc]=16; g_c[cc]=ch-64; g_a0[cc]=0; g_a1[cc]=2; g_av[cc]=1; }
        else              { g_plane[cc]=plane2; g_line[cc]=line2; g_C[cc]=16; g_c[cc]=ch-80; g_a0[cc]=1; g_a1[cc]=2; g_av[cc]=0; }
    }

    const int n_tiles     = (n_points + 15) >> 4;
    const int total_waves = gridDim.x * WAVES_PB;
    const int trips       = (n_tiles + total_waves - 1) / total_waves;   // uniform per block
    const int wave_global = blockIdx.x * WAVES_PB + wv;

    for (int t = 0; t < trips; ++t) {
        const int tile   = wave_global + t * total_waves;
        const bool active = tile < n_tiles;

        // ---- Phase 1: per-point coordinates (lanes 0-15) ----
        if (active && lane < 16) {
            size_t pidx = (size_t)tile * 16 + lane;
            if (pidx >= (size_t)n_points) pidx = (size_t)n_points - 1;  // clamp, stores guarded later
            #pragma unroll
            for (int a = 0; a < 3; ++a) {
                float pt = pts[pidx * 3 + a];
                float x  = (pt - AABB_MIN) * AABB_INV * (float)(GRID_SZ - 1);
                float xf = floorf(x);
                xf = fminf(fmaxf(xf, 0.0f), (float)(GRID_SZ - 1));
                int x0 = (int)xf;
                int x1 = min(x0 + 1, GRID_SZ - 1);
                s_i0[wv][lane][a] = x0;
                s_i1[wv][lane][a] = x1;
                s_w [wv][lane][a] = x - xf;
            }
            // warm next tile's points (global_prefetch_b8)
            size_t nidx = (size_t)(tile + total_waves) * 16 + lane;
            if (nidx < (size_t)n_points)
                __builtin_prefetch(&pts[nidx * 3], 0, 3);
        }
        __syncthreads();

        // ---- Phase 2: coalesced channel-parallel gathers -> fused f16 feature ----
        if (active) {
            #pragma unroll
            for (int cc = 0; cc < 3; ++cc) {
                const float* __restrict__ pl = g_plane[cc];
                const float* __restrict__ ln = g_line[cc];
                const int C = g_C[cc], c = g_c[cc];
                const int a0 = g_a0[cc], a1 = g_a1[cc], av = g_av[cc];
                const int ch = cc * 32 + lane;
                for (int p = 0; p < 16; ++p) {
                    const int   x0 = s_i0[wv][p][a0], x1 = s_i1[wv][p][a0];
                    const int   y0 = s_i0[wv][p][a1], y1 = s_i1[wv][p][a1];
                    const int   v0 = s_i0[wv][p][av], v1 = s_i1[wv][p][av];
                    const float wx = s_w[wv][p][a0];
                    const float wy = s_w[wv][p][a1];
                    const float wl = s_w[wv][p][av];
                    // coalesced: lanes cover contiguous channels c
                    const float p00 = pl[((size_t)y0 * GRID_SZ + x0) * C + c];
                    const float p01 = pl[((size_t)y0 * GRID_SZ + x1) * C + c];
                    const float p10 = pl[((size_t)y1 * GRID_SZ + x0) * C + c];
                    const float p11 = pl[((size_t)y1 * GRID_SZ + x1) * C + c];
                    const float l0  = ln[(size_t)v0 * C + c];
                    const float l1  = ln[(size_t)v1 * C + c];
                    const float top = p00 + wx * (p01 - p00);
                    const float bot = p10 + wx * (p11 - p10);
                    const float pv  = top + wy * (bot - top);
                    const float lv  = l0  + wl * (l1  - l0);
                    s_feat[wv][p][ch] = (_Float16)(pv * lv);
                }
            }
        }
        __syncthreads();

        // ---- Phase 3: WMMA  D(16x32) = A(16x96) x B(96x32), f32 accumulate ----
        v8f acc0 = {};
        v8f acc1 = {};
        const int m_row = lane & 15;
        #pragma unroll
        for (int kc = 0; kc < 3; ++kc) {
            v16h a;
            #pragma unroll
            for (int j = 0; j < 16; ++j)
                a[j] = s_feat[wv][m_row][kc * 32 + kb_idx(lane, j)];
            acc0 = __builtin_amdgcn_wmma_f32_16x16x32_f16(
                false, a, false, bmat[kc][0], (short)0, acc0, false, false);
            acc1 = __builtin_amdgcn_wmma_f32_16x16x32_f16(
                false, a, false, bmat[kc][1], (short)0, acc1, false, false);
        }

        // ---- Store D: VGPR r -> M = r + 8*(lane>=16), N = lane%16 ----
        // Fast path: full tile in-bounds -> 16 unguarded NT stores off one base
        // (immediate offsets, clause-able). NT keeps the streamed output from
        // evicting the L2-resident planes.
        const int mbase = ((lane >> 4) & 1) * 8;
        const size_t row0 = (size_t)tile * 16;
        if (row0 + 16 <= (size_t)n_points) {
            float* __restrict__ outp = out + (row0 + mbase) * APP_DIM + n_col;
            #pragma unroll
            for (int r = 0; r < 8; ++r) {
                __builtin_nontemporal_store(acc0[r], outp + (size_t)r * APP_DIM);
                __builtin_nontemporal_store(acc1[r], outp + (size_t)r * APP_DIM + 16);
            }
        } else if (row0 < (size_t)n_points) {
            #pragma unroll
            for (int r = 0; r < 8; ++r) {
                size_t row = row0 + mbase + r;
                if (row < (size_t)n_points) {
                    __builtin_nontemporal_store(acc0[r], out + row * APP_DIM + n_col);
                    __builtin_nontemporal_store(acc1[r], out + row * APP_DIM + 16 + n_col);
                }
            }
        }
        __syncthreads();   // protect LDS feat before next trip's writes
    }
}

extern "C" void kernel_launch(void* const* d_in, const int* in_sizes, int n_in,
                              void* d_out, int out_size, void* d_ws, size_t ws_size,
                              hipStream_t stream) {
    const float* pts     = (const float*)d_in[0];
    const float* plane0  = (const float*)d_in[1];
    const float* plane1  = (const float*)d_in[2];
    const float* plane2  = (const float*)d_in[3];
    const float* line0   = (const float*)d_in[4];
    const float* line1   = (const float*)d_in[5];
    const float* line2   = (const float*)d_in[6];
    const float* basis_W = (const float*)d_in[7];
    float* out = (float*)d_out;

    const int n_points = in_sizes[0] / 3;     // 524288
    dim3 block(256);                          // 8 wave32
    dim3 grid(512);                           // 4096 waves -> 8 tiles/wave for N=524288
    tensorf_app_kernel<<<grid, block, 0, stream>>>(
        pts, plane0, plane1, plane2, line0, line1, line2, basis_W, out, n_points);
}
